// DualVQQuantizer_53266184405017
// MI455X (gfx1250) — matrix operand order repaired and checked by reading, compiled, not run on hardware
//
#include <hip/hip_runtime.h>
#include <hip/hip_bf16.h>

typedef __attribute__((ext_vector_type(16))) _Float16 v16h;
typedef __attribute__((ext_vector_type(8)))  _Float16 v8h;
typedef __attribute__((ext_vector_type(8)))  float    v8f;

#define B_N 65536
#define D_N 512
#define K_N 1024
#define H_N 512

__device__ __forceinline__ float wave_reduce_add(float v) {
#pragma unroll
  for (int o = 16; o > 0; o >>= 1) v += __shfl_down(v, o, 32);
  return v;
}

// ---------------------------------------------------------------------------
// VQ distance-argmin via split-precision f16 WMMA.
// Block = 128 threads (4 waves). Block owns 64 rows of h; each wave a 16-row
// slab. h staged in LDS as f16 hi/lo; codebook staged 16 columns at a time.
// dist = |h|^2 + |c|^2 - 2 h.c ; argmin over K with running per-lane min.
// ---------------------------------------------------------------------------
__global__ __launch_bounds__(128)
void vq_argmin_kernel(const float* __restrict__ h,
                      const float* __restrict__ cb,
                      int*   __restrict__ idxOut,
                      float* __restrict__ sumOut,
                      float* __restrict__ idxOutF)
{
  extern __shared__ char smem[];
  _Float16* Ah = (_Float16*)smem;          // [64][512] hi
  _Float16* Al = Ah + 64 * D_N;            // [64][512] lo
  _Float16* Bh = Al + 64 * D_N;            // [16][512] hi
  _Float16* Bl = Bh + 16 * D_N;            // [16][512] lo
  float* hsq   = (float*)(Bl + 16 * D_N);  // [64]
  float* cbsq  = hsq + 64;                 // [16]

  const int tid  = threadIdx.x;
  const int lane = tid & 31;
  const int wave = tid >> 5;
  const int rowBase = blockIdx.x * 64;

  // ---- stage A: 64 rows of h -> f16 hi/lo, and per-row |h|^2 ----
  {
    const int r = tid >> 1;
    const int part = tid & 1;
    const float* src = h + (size_t)(rowBase + r) * D_N + part * 256;
    _Float16* dh = Ah + r * D_N + part * 256;
    _Float16* dl = Al + r * D_N + part * 256;
    float sq = 0.f;
#pragma unroll 4
    for (int i = 0; i < 64; ++i) {
      float4 v = *(const float4*)(src + i * 4);
      float xs[4] = {v.x, v.y, v.z, v.w};
#pragma unroll
      for (int k = 0; k < 4; ++k) {
        float x = xs[k];
        _Float16 hi = (_Float16)x;
        _Float16 lo = (_Float16)(x - (float)hi);
        dh[i * 4 + k] = hi;
        dl[i * 4 + k] = lo;
        sq = fmaf(x, x, sq);
      }
    }
    sq += __shfl_xor(sq, 1, 32);
    if (part == 0) hsq[r] = sq;
  }

  float rmin[8];
  int   ridx[8];
#pragma unroll
  for (int j = 0; j < 8; ++j) { rmin[j] = 3.4e38f; ridx[j] = 0; }

  const bool hiHalf = lane >= 16;
  const int  mrowL  = wave * 16 + (lane & 15);   // A row in LDS
  const int  kOffA  = hiHalf ? 8 : 0;            // A frag K offset per ISA layout
  const int  ncol   = lane & 15;                 // B/C column
  const int  kOffB  = hiHalf ? 8 : 0;            // B frag K offset per ISA layout
  const _Float16* arowH = Ah + mrowL * D_N;
  const _Float16* arowL = Al + mrowL * D_N;
  const _Float16* browH = Bh + ncol * D_N;
  const _Float16* browL = Bl + ncol * D_N;

  for (int tile = 0; tile < K_N / 16; ++tile) {
    __syncthreads();  // A ready / previous compute done
    // ---- stage B: 16 codebook rows -> f16 hi/lo, and per-row |c|^2 ----
    {
      const int rc = tid >> 3, c8 = tid & 7;
      const float* src = cb + (size_t)(tile * 16 + rc) * D_N + c8 * 64;
      _Float16* dh = Bh + rc * D_N + c8 * 64;
      _Float16* dl = Bl + rc * D_N + c8 * 64;
      float sq = 0.f;
#pragma unroll 4
      for (int i = 0; i < 16; ++i) {
        float4 v = *(const float4*)(src + i * 4);
        float xs[4] = {v.x, v.y, v.z, v.w};
#pragma unroll
        for (int k = 0; k < 4; ++k) {
          float x = xs[k];
          _Float16 hi = (_Float16)x;
          _Float16 lo = (_Float16)(x - (float)hi);
          dh[i * 4 + k] = hi;
          dl[i * 4 + k] = lo;
          sq = fmaf(x, x, sq);
        }
      }
      sq += __shfl_xor(sq, 1, 32);
      sq += __shfl_xor(sq, 2, 32);
      sq += __shfl_xor(sq, 4, 32);
      if (c8 == 0) cbsq[rc] = sq;
    }
    __syncthreads();

    // ---- 16x16 dot-product accumulation over full D ----
    v8f c = {};
#pragma unroll 2
    for (int ds = 0; ds < D_N / 32; ++ds) {
      const int d0 = ds * 32;
      v8h a0  = *(const v8h*)(arowH + d0 + kOffA);
      v8h a1  = *(const v8h*)(arowH + d0 + 16 + kOffA);
      v8h la0 = *(const v8h*)(arowL + d0 + kOffA);
      v8h la1 = *(const v8h*)(arowL + d0 + 16 + kOffA);
      v8h b0  = *(const v8h*)(browH + d0 + kOffB);
      v8h b1  = *(const v8h*)(browH + d0 + 16 + kOffB);
      v8h lb0 = *(const v8h*)(browL + d0 + kOffB);
      v8h lb1 = *(const v8h*)(browL + d0 + 16 + kOffB);
      v16h ah = __builtin_shufflevector(a0,  a1,  0,1,2,3,4,5,6,7,8,9,10,11,12,13,14,15);
      v16h al = __builtin_shufflevector(la0, la1, 0,1,2,3,4,5,6,7,8,9,10,11,12,13,14,15);
      v16h bh = __builtin_shufflevector(b0,  b1,  0,1,2,3,4,5,6,7,8,9,10,11,12,13,14,15);
      v16h bl = __builtin_shufflevector(lb0, lb1, 0,1,2,3,4,5,6,7,8,9,10,11,12,13,14,15);
      // split-precision: hi*hi + hi*lo + lo*hi  (~f32 accuracy)
      c = __builtin_amdgcn_wmma_f32_16x16x32_f16(false, ah, false, bh, (short)0, c, false, false);
      c = __builtin_amdgcn_wmma_f32_16x16x32_f16(false, ah, false, bl, (short)0, c, false, false);
      c = __builtin_amdgcn_wmma_f32_16x16x32_f16(false, al, false, bh, (short)0, c, false, false);
    }

    const float cs = cbsq[ncol];
    const int ci = tile * 16 + ncol;
#pragma unroll
    for (int j = 0; j < 8; ++j) {
      float val = fmaf(-2.0f, c[j], cs);   // |c|^2 - 2 h.c  (|h|^2 added later)
      if (val < rmin[j]) { rmin[j] = val; ridx[j] = ci; }
    }
  }

  // ---- argmin reduce across the 16 lanes sharing the same row set ----
#pragma unroll
  for (int m = 1; m < 16; m <<= 1) {
#pragma unroll
    for (int j = 0; j < 8; ++j) {
      float ov = __shfl_xor(rmin[j], m, 32);
      int   oi = __shfl_xor(ridx[j], m, 32);
      if (ov < rmin[j] || (ov == rmin[j] && oi < ridx[j])) { rmin[j] = ov; ridx[j] = oi; }
    }
  }
  if ((lane & 15) == 0) {
    const int rloc = wave * 16 + (hiHalf ? 8 : 0);
    float s = 0.f;
#pragma unroll
    for (int j = 0; j < 8; ++j) {
      const int lrow = rloc + j;
      const int row  = rowBase + lrow;
      idxOut[row]  = ridx[j];
      idxOutF[row] = (float)ridx[j];
      s += rmin[j] + hsq[lrow];            // min squared distance = sum diff^2
    }
    atomicAdd(sumOut, s);
  }
}

// ---------------------------------------------------------------------------
__global__ void one_hot_kernel(const int* __restrict__ idx, float* __restrict__ q) {
  size_t i = (size_t)blockIdx.x * blockDim.x + threadIdx.x;  // over B*K/4
  int b  = (int)(i >> 8);        // K/4 = 256
  int k0 = ((int)i & 255) * 4;
  int id = idx[b];
  float4 v;
  v.x = (k0     == id) ? 1.0f : 0.0f;
  v.y = (k0 + 1 == id) ? 1.0f : 0.0f;
  v.z = (k0 + 2 == id) ? 1.0f : 0.0f;
  v.w = (k0 + 3 == id) ? 1.0f : 0.0f;
  *(float4*)(q + i * 4) = v;
}

__global__ void gather3_kernel(const int* __restrict__ idx, const float* __restrict__ cb,
                               float* __restrict__ s, float* __restrict__ hd,
                               float* __restrict__ qt) {
  size_t i = (size_t)blockIdx.x * blockDim.x + threadIdx.x;  // over B*D/4
  int b   = (int)(i >> 7);       // D/4 = 128
  int off = ((int)i & 127) * 4;
  float4 v = *(const float4*)(cb + (size_t)idx[b] * D_N + off);
  *(float4*)(s  + i * 4) = v;
  *(float4*)(hd + i * 4) = v;
  *(float4*)(qt + i * 4) = v;
}

// ---------------------------------------------------------------------------
// Coupling table: for each code j, L[j][n] = 0.5*(silu(W1[:,j]+b1) . W2[n] + b2[n])
// and lse[j] = logsumexp_n L[j][n].
// ---------------------------------------------------------------------------
__global__ __launch_bounds__(256)
void lmat_kernel(const float* __restrict__ W1, const float* __restrict__ b1,
                 const float* __restrict__ W2, const float* __restrict__ b2,
                 float* __restrict__ Lmat, float* __restrict__ lse)
{
  __shared__ float hdn[H_N];
  __shared__ float red[256];
  const int j = blockIdx.x;
  const int t = threadIdx.x;
  for (int s = 0; s < H_N; s += 256) {
    int hh = s + t;
    float x = W1[(size_t)hh * K_N + j] + b1[hh];
    hdn[hh] = x / (1.0f + expf(-x));     // silu
  }
  __syncthreads();

  float vals[4];
#pragma unroll
  for (int s = 0; s < 4; ++s) {
    const int n = t + s * 256;
    float acc = b2[n];
    const float* w2r = W2 + (size_t)n * H_N;
#pragma unroll 8
    for (int hh = 0; hh < H_N; ++hh) acc = fmaf(hdn[hh], w2r[hh], acc);
    float hv = 0.5f * acc;               // logits / TR_TEMP(=2)
    vals[s] = hv;
    Lmat[(size_t)j * K_N + n] = hv;
  }
  float mx = fmaxf(fmaxf(vals[0], vals[1]), fmaxf(vals[2], vals[3]));
  red[t] = mx; __syncthreads();
  for (int o = 128; o > 0; o >>= 1) { if (t < o) red[t] = fmaxf(red[t], red[t + o]); __syncthreads(); }
  mx = red[0]; __syncthreads();
  float sm = 0.f;
#pragma unroll
  for (int s = 0; s < 4; ++s) sm += expf(vals[s] - mx);
  red[t] = sm; __syncthreads();
  for (int o = 128; o > 0; o >>= 1) { if (t < o) red[t] += red[t + o]; __syncthreads(); }
  if (t == 0) lse[j] = mx + logf(red[0]);
}

__global__ void kl_kernel(const float* __restrict__ Lmat, const float* __restrict__ lse,
                          const int* __restrict__ idxTr, const int* __restrict__ idxRe,
                          float* __restrict__ acc) {
  int b = blockIdx.x * blockDim.x + threadIdx.x;
  float v = 0.f;
  if (b < B_N) {
    int jr = idxRe[b], jt = idxTr[b];
    v = Lmat[(size_t)jr * K_N + jt] - lse[jr];   // logp[b, idx_tr[b]]
  }
  v = wave_reduce_add(v);
  if ((threadIdx.x & 31) == 0) atomicAdd(acc, v);
}

__global__ void mask_kernel(const float* __restrict__ m, float* __restrict__ acc) {
  size_t i = (size_t)blockIdx.x * blockDim.x + threadIdx.x;
  float v = fabsf(m[i]);
  v = wave_reduce_add(v);
  if ((threadIdx.x & 31) == 0) atomicAdd(acc, v);
}

__global__ void finalize_kernel(const float* __restrict__ acc, float* __restrict__ lossOut) {
  float vq = 1.25f * (acc[0] + acc[1]) / (float)((size_t)B_N * D_N);  // (1+beta)*mean
  float kl = -acc[2] / (float)B_N;
  float sp = acc[3] / (float)((size_t)H_N * K_N);
  lossOut[0] = vq + 0.1f * kl + 0.1f * sp;
}

// ---------------------------------------------------------------------------
extern "C" void kernel_launch(void* const* d_in, const int* in_sizes, int n_in,
                              void* d_out, int out_size, void* d_ws, size_t ws_size,
                              hipStream_t stream) {
  (void)in_sizes; (void)n_in; (void)out_size; (void)ws_size;
  const float* h_tr  = (const float*)d_in[0];
  const float* h_re  = (const float*)d_in[1];
  const float* cb_tr = (const float*)d_in[2];
  const float* cb_re = (const float*)d_in[3];
  const float* W1    = (const float*)d_in[4];
  const float* b1    = (const float*)d_in[5];
  const float* W2    = (const float*)d_in[6];
  const float* b2    = (const float*)d_in[7];
  const float* mask  = (const float*)d_in[8];

  float* out = (float*)d_out;
  const size_t BK = (size_t)B_N * K_N, BD = (size_t)B_N * D_N;
  float* q_tr     = out;
  float* soft_tr  = q_tr + BK;
  float* hard_tr  = soft_tr + BD;
  float* quant_tr = hard_tr + BD;
  float* idxf_tr  = quant_tr + BD;
  float* q_re     = idxf_tr + B_N;
  float* soft_re  = q_re + BK;
  float* hard_re  = soft_re + BD;
  float* quant_re = hard_re + BD;
  float* idxf_re  = quant_re + BD;
  float* lossOut  = idxf_re + B_N;

  int*   idxTr = (int*)d_ws;
  int*   idxRe = idxTr + B_N;
  float* acc   = (float*)(idxRe + B_N);          // [4] sumTr,sumRe,kl,mask
  float* Lmat  = acc + 4;                         // [K][K]
  float* lse   = Lmat + (size_t)K_N * K_N;        // [K]

  hipMemsetAsync(acc, 0, 4 * sizeof(float), stream);

  const int SMEM = (64 * D_N + 64 * D_N + 16 * D_N + 16 * D_N) * 2 + (64 + 16) * 4;
  hipFuncSetAttribute(reinterpret_cast<const void*>(vq_argmin_kernel),
                      hipFuncAttributeMaxDynamicSharedMemorySize, SMEM);

  vq_argmin_kernel<<<B_N / 64, 128, SMEM, stream>>>(h_tr, cb_tr, idxTr, acc + 0, idxf_tr);
  vq_argmin_kernel<<<B_N / 64, 128, SMEM, stream>>>(h_re, cb_re, idxRe, acc + 1, idxf_re);

  one_hot_kernel<<<(unsigned)((BK / 4) / 256), 256, 0, stream>>>(idxTr, q_tr);
  one_hot_kernel<<<(unsigned)((BK / 4) / 256), 256, 0, stream>>>(idxRe, q_re);

  gather3_kernel<<<(unsigned)((BD / 4) / 256), 256, 0, stream>>>(idxTr, cb_tr, soft_tr, hard_tr, quant_tr);
  gather3_kernel<<<(unsigned)((BD / 4) / 256), 256, 0, stream>>>(idxRe, cb_re, soft_re, hard_re, quant_re);

  lmat_kernel<<<K_N, 256, 0, stream>>>(W1, b1, W2, b2, Lmat, lse);
  kl_kernel<<<B_N / 256, 256, 0, stream>>>(Lmat, lse, idxTr, idxRe, acc + 2);
  mask_kernel<<<(H_N * K_N) / 256, 256, 0, stream>>>(mask, acc + 3);
  finalize_kernel<<<1, 1, 0, stream>>>(acc, lossOut);
}
